// NewAttention_15556371546524
// MI455X (gfx1250) — matrix operand order repaired and checked by reading, not compiled, not run on hardware
//
#include <hip/hip_runtime.h>
#include <hip/hip_fp16.h>

typedef _Float16 f16;
typedef __attribute__((ext_vector_type(16))) _Float16 v16h;
typedef __attribute__((ext_vector_type(8)))  float    v8f;

#define B_    8
#define N_    1024
#define NM_   1023      // n - NUM_CLS
#define DIMC  512
#define HC    8
#define DHC   64
#define INNER 512
#define M_TOT (B_*N_)   // 8192 rows for the big GEMMs

// LDS layout for k_attn (padded row strides -> bank-conflict-free b128 reads)
#define LSTR 1028                   // logits row stride (floats);  1028%64? -> 4-bank step
#define ASTR 1032                   // attn f16 row stride; 516 words -> 4-bank step, 16B aligned
#define QSTR 72                     // staged-Q row stride (f16); 36 words -> distinct banks
#define SM_L_BYTES (16 * LSTR * 4)  // 65792
#define SM_A_BYTES (16 * ASTR * 2)  // 33024
#define SM_Q_BYTES (16 * QSTR * 2)  // 2304
#define SM_TOTAL   (SM_L_BYTES + SM_A_BYTES + SM_Q_BYTES)  // 101120 <= 320KB/WGP

// ---- fragment loaders -------------------------------------------------------
// A-fragment (16x32 f16): lane L: m = L&15, half = L>>4; v[0..7] = A[m][k0+half*8 ..],
// v[8..15] = A[m][k0+16+half*8 ..]  -> two 16B contiguous chunks.
__device__ __forceinline__ v16h ld8x2(const f16* p0, const f16* p1) {
  union { uint4 u[2]; v16h v; } c;
  c.u[0] = *(const uint4*)p0;
  c.u[1] = *(const uint4*)p1;
  return c.v;
}
// B-fragment (32x16 f16): lane L: n = L&15, half = L>>4; same k order as A (ld8x2).
// With B stored transposed (row n contiguous in k) this is two 16B chunks (ld8x2); ld16 is unused.
__device__ __forceinline__ v16h ld16(const f16* p) {
  union { uint4 u[2]; v16h v; } c;
  c.u[0] = ((const uint4*)p)[0];
  c.u[1] = ((const uint4*)p)[1];
  return c.v;
}
__device__ __forceinline__ v8f wmma_f16(v16h a, v16h b, v8f c) {
  // (neg_a, A, neg_b, B, c_mod, C, reuse_a, reuse_b)
  return __builtin_amdgcn_wmma_f32_16x16x32_f16(false, a, false, b, (short)0, c,
                                                false, false);
}
__device__ __forceinline__ float wave_max(float v) {
  for (int o = 16; o >= 1; o >>= 1) v = fmaxf(v, __shfl_xor(v, o, 32));
  return v;
}
__device__ __forceinline__ float wave_sum(float v) {
  for (int o = 16; o >= 1; o >>= 1) v += __shfl_xor(v, o, 32);
  return v;
}

// ---- prep: fp32 -> f16, and weight transpose -------------------------------
__global__ void k_convert(const float* __restrict__ src, f16* __restrict__ dst, int n) {
  int i = blockIdx.x * 256 + threadIdx.x;
  if (i < n) dst[i] = (f16)src[i];
}
// WT[n][k] = W[k][n], 512x512
__global__ void k_transpose(const float* __restrict__ W, f16* __restrict__ WT) {
  int i = blockIdx.x * 256 + threadIdx.x;   // 512*512 threads
  int n = i >> 9, k = i & 511;
  WT[n * 512 + k] = (f16)W[k * 512 + n];
}

// ---- QKV projection: one wave computes one 16x16 tile, K-loop over DIM -----
// mode z: 0 -> Q (scaled by DH^-0.5, layout [bh][i][d])
//         1 -> K (layout [bh][j][d])
//         2 -> V (transposed layout [bh][d][j])
__global__ void __launch_bounds__(32)
k_qkv(const f16* __restrict__ X,
      const f16* __restrict__ WqT, const f16* __restrict__ WkT, const f16* __restrict__ WvT,
      f16* __restrict__ Q, f16* __restrict__ K, f16* __restrict__ Vt) {
  const int lane = threadIdx.x;
  const int half = lane >> 4, sub = lane & 15;
  const int n0 = blockIdx.x * 16;
  const int m0 = blockIdx.y * 16;
  const int mode = blockIdx.z;
  const f16* WT = (mode == 0) ? WqT : ((mode == 1) ? WkT : WvT);

  const f16* arow = X  + (size_t)(m0 + sub) * DIMC;
  const f16* brow = WT + (size_t)(n0 + sub) * DIMC;
  v8f acc = {};
  for (int k0 = 0; k0 < DIMC; k0 += 32) {
    if (k0 + 64 < DIMC) {
      __builtin_prefetch(arow + k0 + 64, 0, 3);   // near-scope prefetch
      __builtin_prefetch(brow + k0 + 64, 0, 3);
    }
    v16h a = ld8x2(arow + k0 + half * 8, arow + k0 + 16 + half * 8);
    v16h b = ld8x2(brow + k0 + half * 8, brow + k0 + 16 + half * 8);
    acc = wmma_f16(a, b, acc);
  }
  const int c = n0 + sub;          // inner-dim column
  const int h = c >> 6, d = c & 63;
  for (int r = 0; r < 8; ++r) {
    int mg = m0 + half * 8 + r;    // global row in [0, 8192)
    int bi = mg >> 10, i = mg & 1023;
    int bh = bi * HC + h;
    float v = acc[r];
    if (mode == 0)      Q [((size_t)bh * N_  + i) * DHC + d] = (f16)(v * 0.125f);
    else if (mode == 1) K [((size_t)bh * N_  + i) * DHC + d] = (f16)v;
    else                Vt[((size_t)bh * DHC + d) * N_  + i] = (f16)v;
  }
}

// ---- fused attention: dots -> mask -> softmax -> attn write -> attn @ V ----
// grid: x = 64 (16-row blocks of i), y = 64 (b*H). 128 threads = 4 waves.
__global__ void __launch_bounds__(128)
k_attn(const f16* __restrict__ Q, const f16* __restrict__ K, const f16* __restrict__ Vt,
       const unsigned char* __restrict__ mask,
       const int* __restrict__ seq_p, const int* __restrict__ layer_p,
       float* __restrict__ attn_f32, f16* __restrict__ AO) {
  extern __shared__ char smem[];
  float* L  = (float*)smem;                                // [16][LSTR] fp32 logits
  f16*   Ah = (f16*)(smem + SM_L_BYTES);                   // [16][ASTR] f16 attn
  f16*   LQ = (f16*)(smem + SM_L_BYTES + SM_A_BYTES);      // [16][QSTR] staged Q panel

  const int tid  = threadIdx.x;
  const int wave = tid >> 5, lane = tid & 31;
  const int half = lane >> 4, sub = lane & 15;
  const int i0 = blockIdx.x * 16;
  const int bh = blockIdx.y;
  const int bi = bh >> 3;
  const int S     = seq_p[0];
  const int layer = layer_p[0];
  const float NEGMAX = -3.402823466e+38f;

  const f16* Qb = Q  + (size_t)bh * N_ * DHC;
  const f16* Kb = K  + (size_t)bh * N_ * DHC;
  const f16* Vb = Vt + (size_t)bh * DHC * N_;
  const unsigned char* mflat = mask + (size_t)bi * NM_;

  auto mi = [&](int t) -> bool { return t == 0 ? true : (mflat[t - 1] != 0); };

  // ---- stage the 16x64 Q panel into LDS once (async copy, shared by 4 waves)
  {
    const int qrow  = tid >> 3;          // 16 rows
    const int chunk = tid & 7;           // 8 x 16B per row
    const f16* src = Qb + (size_t)(i0 + qrow) * DHC + chunk * 8;
    unsigned  dst = (unsigned)(size_t)(LQ + qrow * QSTR + chunk * 8);
    asm volatile("global_load_async_to_lds_b128 %0, %1, off"
                 :: "v"(dst), "v"(src) : "memory");
    asm volatile("s_wait_asynccnt 0x0" ::: "memory");
  }

  // per-lane row properties for the 8 D-rows it owns
  bool mr[8], ar[8];
  for (int r = 0; r < 8; ++r) {
    int ig = i0 + half * 8 + r;
    mr[r] = mi(ig);
    ar[r] = (ig >= 1) && (ig - 1 < S);
  }
  __syncthreads();   // Q panel visible to all waves

  // Q fragments from LDS (ds_load_b128), reused for all 16 j-tiles of this wave
  const f16* qlds = LQ + sub * QSTR;
  v16h a0 = ld8x2(qlds +      half * 8, qlds + 16 + half * 8);
  v16h a1 = ld8x2(qlds + 32 + half * 8, qlds + 48 + half * 8);

  // ---- phase 1: logits (Q K^T), mask, into LDS ----
  for (int jt = wave * 16; jt < wave * 16 + 16; ++jt) {
    const int j0 = jt * 16;
    const f16* krow = Kb + (size_t)(j0 + sub) * DHC;
    v16h b0 = ld8x2(krow +      half * 8, krow + 16 + half * 8);
    v16h b1 = ld8x2(krow + 32 + half * 8, krow + 48 + half * 8);
    v8f acc = {};
    acc = wmma_f16(a0, b0, acc);
    acc = wmma_f16(a1, b1, acc);

    const int jg = j0 + sub;
    const bool mj = mi(jg);
    const bool aj = (jg >= 1) && (jg - 1 < S);
    for (int r = 0; r < 8; ++r) {
      const int ig = i0 + half * 8 + r;
      bool ok;
      if (layer == 0) {
        if (ig == 0)      ok = mj;
        else if (jg == 0) ok = mr[r];
        else              ok = mr[r] && mj && (ar[r] == aj);
      } else if (layer == 1) {
        bool vi = ig >= S + 1, vj = jg >= S + 1;
        ok = mr[r] && mj && !((ar[r] && aj) || (vi && vj));
      } else if (layer == 2) {
        ok = mr[r] && mj;
      } else ok = true;
      L[(half * 8 + r) * LSTR + jg] = ok ? acc[r] : NEGMAX;
    }
  }
  __syncthreads();

  // ---- phase 2: softmax, one wave per row (4 rows each); 128B-coalesced
  //      attn stores, conflict-free LDS, shfl reductions ----
  for (int rr = 0; rr < 4; ++rr) {
    const int row = wave * 4 + rr;
    float* Lr = L + row * LSTR;
    float lmax = NEGMAX;
    for (int j = lane; j < N_; j += 32) lmax = fmaxf(lmax, Lr[j]);
    const float rmax = wave_max(lmax);
    float lsum = 0.f;
    for (int j = lane; j < N_; j += 32) {
      float e = __expf(Lr[j] - rmax);
      Lr[j] = e;
      lsum += e;
    }
    const float rinv = 1.0f / wave_sum(lsum);
    float* outrow = attn_f32 + ((size_t)bh * N_ + (i0 + row)) * N_;
    f16*   ahrow  = Ah + row * ASTR;
    for (int j = lane; j < N_; j += 32) {
      float a = Lr[j] * rinv;
      outrow[j] = a;           // attn output (fp32, wave-contiguous stores)
      ahrow[j]  = (f16)a;      // feed for attn @ V
    }
  }
  __syncthreads();

  // ---- phase 3: out = attn @ V (each wave: one 16x16 d-tile) ----
  const int d0 = wave * 16;
  const f16* arow2 = Ah + sub * ASTR;               // A row m = sub (padded stride)
  const f16* vrow  = Vb + (size_t)(d0 + sub) * N_;  // B column n = sub (Vt row)
  v8f acc = {};
  for (int k0 = 0; k0 < N_; k0 += 32) {
    v16h a = ld8x2(arow2 + k0 + half * 8, arow2 + k0 + 16 + half * 8); // ds_load_b128
    v16h b = ld8x2(vrow + k0 + half * 8, vrow + k0 + 16 + half * 8);
    acc = wmma_f16(a, b, acc);
  }
  const int h = bh & 7;
  const int col = h * DHC + d0 + sub;
  for (int r = 0; r < 8; ++r) {
    int ig = i0 + half * 8 + r;
    size_t m = (size_t)bi * N_ + ig;
    AO[m * INNER + col] = (f16)acc[r];
  }
}

// ---- out-projection: out = AO @ Wo + bo ------------------------------------
__global__ void __launch_bounds__(32)
k_outproj(const f16* __restrict__ AO, const f16* __restrict__ WoT,
          const float* __restrict__ bo, float* __restrict__ out) {
  const int lane = threadIdx.x;
  const int half = lane >> 4, sub = lane & 15;
  const int n0 = blockIdx.x * 16;
  const int m0 = blockIdx.y * 16;
  const f16* arow = AO  + (size_t)(m0 + sub) * INNER;
  const f16* brow = WoT + (size_t)(n0 + sub) * INNER;
  v8f acc = {};
  for (int k0 = 0; k0 < INNER; k0 += 32) {
    if (k0 + 64 < INNER) {
      __builtin_prefetch(arow + k0 + 64, 0, 3);
      __builtin_prefetch(brow + k0 + 64, 0, 3);
    }
    v16h a = ld8x2(arow + k0 + half * 8, arow + k0 + 16 + half * 8);
    v16h b = ld8x2(brow + k0 + half * 8, brow + k0 + 16 + half * 8);
    acc = wmma_f16(a, b, acc);
  }
  const int c = n0 + sub;
  const float bias = bo[c];
  for (int r = 0; r < 8; ++r) {
    int mg = m0 + half * 8 + r;
    out[(size_t)mg * DIMC + c] = acc[r] + bias;
  }
}

extern "C" void kernel_launch(void* const* d_in, const int* in_sizes, int n_in,
                              void* d_out, int out_size, void* d_ws, size_t ws_size,
                              hipStream_t stream) {
  (void)in_sizes; (void)n_in; (void)out_size; (void)ws_size;
  const float*         x      = (const float*)d_in[0];
  const unsigned char* mask   = (const unsigned char*)d_in[1];
  const int*           seq_p  = (const int*)d_in[2];
  const int*           layer_p= (const int*)d_in[3];
  const float*         Wq     = (const float*)d_in[4];
  const float*         Wk     = (const float*)d_in[5];
  const float*         Wv     = (const float*)d_in[6];
  const float*         Wo     = (const float*)d_in[7];
  const float*         bo     = (const float*)d_in[8];

  float* out  = (float*)d_out;                       // [8, 1024, 512]
  float* attn = out + (size_t)M_TOT * DIMC;          // [8, 8, 1024, 1024]

  // workspace layout (f16 = 2 bytes)
  char* ws = (char*)d_ws;
  f16* xb  = (f16*)(ws);                             // 8192*512      -> 8 MB
  f16* WqT = (f16*)(ws + 8388608);                   // 512*512       -> 512 KB
  f16* WkT = (f16*)(ws + 8912896);
  f16* WvT = (f16*)(ws + 9437184);
  f16* WoT = (f16*)(ws + 9961472);
  f16* Qb  = (f16*)(ws + 10485760);                  // [64][1024][64] -> 8 MB
  f16* Kb  = (f16*)(ws + 18874368);                  // [64][1024][64]
  f16* Vtb = (f16*)(ws + 27262976);                  // [64][64][1024]
  f16* AOb = (f16*)(ws + 35651584);                  // [8192][512]

  k_convert<<<(M_TOT * DIMC + 255) / 256, 256, 0, stream>>>(x, xb, M_TOT * DIMC);
  k_transpose<<<1024, 256, 0, stream>>>(Wq, WqT);
  k_transpose<<<1024, 256, 0, stream>>>(Wk, WkT);
  k_transpose<<<1024, 256, 0, stream>>>(Wv, WvT);
  k_transpose<<<1024, 256, 0, stream>>>(Wo, WoT);

  k_qkv<<<dim3(32, 512, 3), 32, 0, stream>>>(xb, WqT, WkT, WvT, Qb, Kb, Vtb);

  k_attn<<<dim3(64, 64), 128, SM_TOTAL, stream>>>(Qb, Kb, Vtb, mask, seq_p, layer_p,
                                                  attn, AOb);

  k_outproj<<<dim3(32, 512), 32, 0, stream>>>(AOb, WoT, bo, out);
}
